// MoEAttention_16423954940129
// MI455X (gfx1250) — matrix operand order repaired and checked by reading, not compiled, not run on hardware
//
#include <hip/hip_runtime.h>
#include <hip/hip_bf16.h>
#include <math.h>

// ---------------- problem constants ----------------
#define NEXP   12
#define TOPK   2
#define NHEAD  16
#define DIM    1024
#define HD     64
#define BATCH  4
#define SEQ    1024
#define T_TOK  (BATCH*SEQ)   // 4096

// ---------------- types ----------------
typedef float  v8f   __attribute__((ext_vector_type(8)));
typedef __bf16 v16bf __attribute__((ext_vector_type(16)));
typedef __bf16 v8bf  __attribute__((ext_vector_type(8)));

__device__ __forceinline__ unsigned short f2bf(float f) {
    unsigned u = __float_as_uint(f);
    unsigned r = (u + 0x7FFFu + ((u >> 16) & 1u)) >> 16;   // round-to-nearest-even
    return (unsigned short)r;
}

__device__ __forceinline__ v16bf cat8(v8bf lo, v8bf hi) {
    return __builtin_shufflevector(lo, hi, 0,1,2,3,4,5,6,7,8,9,10,11,12,13,14,15);
}

__device__ __forceinline__ v8f wmma_bf16(v16bf a, v16bf b, v8f c) {
    return __builtin_amdgcn_wmma_f32_16x16x32_bf16(false, a, false, b, (short)0, c, false, false);
}

__device__ __forceinline__ v16bf splat16(__bf16 w) {
    v16bf r;
#pragma unroll
    for (int j = 0; j < 16; ++j) r[j] = w;
    return r;
}

// ---------------- fp32 -> bf16 convert ----------------
__global__ void cvt_bf16_kernel(const float* __restrict__ src,
                                unsigned short* __restrict__ dst, int n) {
    int i = blockIdx.x * blockDim.x + threadIdx.x;
    int stride = gridDim.x * blockDim.x;
    for (; i < n; i += stride) dst[i] = f2bf(src[i]);
}

// ---------------- router: softmax -> top2 -> renorm ----------------
__global__ __launch_bounds__(128) void router_kernel(
    const float* __restrict__ x, const float* __restrict__ Wr,
    float* __restrict__ wts, float* __restrict__ probs) {
    int lane = threadIdx.x & 31;
    int wid  = threadIdx.x >> 5;
    int t    = blockIdx.x * 4 + wid;
    if (t >= T_TOK) return;

    float logit[NEXP];
    const float* xr = x + (size_t)t * DIM;
    for (int e = 0; e < NEXP; ++e) {
        const float* wr = Wr + (size_t)e * DIM;
        float s = 0.f;
        for (int d = lane; d < DIM; d += 32) s += xr[d] * wr[d];
        for (int off = 16; off; off >>= 1) s += __shfl_xor(s, off, 32);
        logit[e] = s;
    }
    if (lane == 0) {
        float mx = logit[0];
        for (int e = 1; e < NEXP; ++e) mx = fmaxf(mx, logit[e]);
        float p[NEXP], den = 0.f;
        for (int e = 0; e < NEXP; ++e) { p[e] = __expf(logit[e] - mx); den += p[e]; }
        float inv = 1.f / den;
        for (int e = 0; e < NEXP; ++e) { p[e] *= inv; probs[t*NEXP + e] = p[e]; }
        int i0 = 0;
        for (int e = 1; e < NEXP; ++e) if (p[e] > p[i0]) i0 = e;
        int i1 = (i0 == 0) ? 1 : 0;
        for (int e = 0; e < NEXP; ++e) if (e != i0 && p[e] > p[i1]) i1 = e;
        float v0 = p[i0], v1 = p[i1];
        float rs = 1.f / (v0 + v1 + 1e-6f);
        for (int e = 0; e < NEXP; ++e) wts[t*NEXP + e] = 0.f;
        wts[t*NEXP + i0] = v0 * rs;
        wts[t*NEXP + i1] = v1 * rs;
    }
}

// ---------------- deterministic LB reductions ----------------
__global__ __launch_bounds__(256) void lb_reduce_kernel(
    const float* __restrict__ probs, const float* __restrict__ wts,
    float* __restrict__ psum, float* __restrict__ cnt) {
    __shared__ float sp[256], sc[256];
    int e = blockIdx.x, tid = threadIdx.x;
    float s = 0.f, c = 0.f;
    for (int t = tid; t < T_TOK; t += 256) {
        s += probs[t*NEXP + e];
        c += (wts[t*NEXP + e] != 0.f) ? 1.f : 0.f;
    }
    sp[tid] = s; sc[tid] = c; __syncthreads();
    for (int off = 128; off; off >>= 1) {
        if (tid < off) { sp[tid] += sp[tid+off]; sc[tid] += sc[tid+off]; }
        __syncthreads();
    }
    if (tid == 0) { psum[e] = sp[0]; cnt[e] = sc[0]; }
}

__global__ void lb_final_kernel(const float* __restrict__ psum,
                                const float* __restrict__ cnt,
                                float* __restrict__ out_lb) {
    float tot = 0.f;
    for (int e = 0; e < NEXP; ++e) tot += cnt[e];
    float lb = 0.f;
    for (int e = 0; e < NEXP; ++e) lb += (cnt[e] / (tot + 1e-6f)) * psum[e];
    out_lb[0] = lb * (float)NEXP;
}

// ---------------- MoE-weighted GEMM: Y[t,o] = sum_e w[t,e] * (Xb @ We^T) ----------------
// One wave computes a 64(M) x 32(N) tile. K outer / experts inner: the 4 A-frags are
// loaded ONCE per K-step and reused across all 12 experts with the router weight
// pre-multiplied into A (w scales rows of A), so a single accumulator set suffices.
// 96 WMMAs per 896B/lane of loads (~55 FLOP/B). out_mode: 0=f32[T,D], 1=bf16[T,D],
// 2=bf16 head-transposed [B*H][HD][SEQ] (for V).
__global__ __launch_bounds__(256) void moe_gemm_kernel(
    const unsigned short* __restrict__ Xb,   // [T,DIM] bf16 bits
    const unsigned short* __restrict__ Wb,   // [NEXP,DIM,DIM] bf16 bits
    const float* __restrict__ wts,           // [T,NEXP]
    void* __restrict__ Yout, int out_mode) {
    int lane = threadIdx.x & 31;
    int wave = blockIdx.x * 8 + (threadIdx.x >> 5);
    int mg = wave >> 5;        // 0..63  (64-row group)
    int nt = wave & 31;        // 0..31  (32-col tile)
    int mb = mg * 64;
    int nb = nt * 32;
    int l15 = lane & 15;
    int hi  = lane >> 4;
    int AkL = hi ? 8 : 0;      // A-frag K half offset (ISA 16-bit A layout)
    int BkL = hi ? 8 : 0;      // B-frag K half offset (ISA 16-bit B layout)

    // Per-lane router weights: A-frag i covers only row mb+16i+l15 in this lane.
    __bf16 wbr[4][NEXP];
#pragma unroll
    for (int i = 0; i < 4; ++i) {
        const float* wr = wts + (size_t)(mb + 16*i + l15) * NEXP;
#pragma unroll
        for (int e = 0; e < NEXP; ++e)
            wbr[i][e] = __builtin_bit_cast(__bf16, f2bf(wr[e]));
    }

    v8f acc[8] = {};           // [0..3]: cols nb..+15, [4..7]: cols nb+16..+31

    const unsigned short* Bbase =
        Wb + (size_t)(nb + l15) * DIM + BkL;          // expert 0, row nb+l15
    const unsigned short* Abase =
        Xb + (size_t)(mb + l15) * DIM + AkL;

#pragma unroll 1
    for (int kb = 0; kb < DIM; kb += 32) {
        // raw A fragments for this K-step (shared across all experts)
        v16bf araw[4];
#pragma unroll
        for (int i = 0; i < 4; ++i) {
            const unsigned short* Ar = Abase + (size_t)(16*i) * DIM + kb;
            araw[i] = cat8(*(const v8bf*)(Ar), *(const v8bf*)(Ar + 16));
        }
#pragma unroll
        for (int e = 0; e < NEXP; ++e) {
            const unsigned short* Bp = Bbase + (size_t)e * DIM * DIM + kb;
            v16bf b0 = cat8(*(const v8bf*)(Bp), *(const v8bf*)(Bp + 16));
            v16bf b1 = cat8(*(const v8bf*)(Bp + (size_t)16 * DIM), *(const v8bf*)(Bp + (size_t)16 * DIM + 16));
#pragma unroll
            for (int i = 0; i < 4; ++i) {
                v16bf as = araw[i] * splat16(wbr[i][e]);
                acc[i]     = wmma_bf16(as, b0, acc[i]);
                acc[4 + i] = wmma_bf16(as, b1, acc[4 + i]);
            }
        }
    }

#pragma unroll
    for (int j = 0; j < 2; ++j) {
        int col = nb + 16*j + l15;
#pragma unroll
        for (int i = 0; i < 4; ++i) {
            int rbase = mb + 16*i + 8*hi;
#pragma unroll
            for (int v = 0; v < 8; ++v) {
                int row = rbase + v;
                float val = acc[4*j + i][v];
                if (out_mode == 0) {
                    ((float*)Yout)[(size_t)row * DIM + col] = val;
                } else if (out_mode == 1) {
                    ((unsigned short*)Yout)[(size_t)row * DIM + col] = f2bf(val);
                } else {
                    int b = row >> 10, n = row & (SEQ-1);
                    int h = col >> 6,  hd = col & (HD-1);
                    ((unsigned short*)Yout)[ (((size_t)(b*NHEAD + h)*HD + hd) << 10) + n ] = f2bf(val);
                }
            }
        }
    }
}

// ---------------- flash attention (non-causal), bf16 WMMA, fp32 softmax ----------------
__global__ __launch_bounds__(128) void attn_kernel(
    const unsigned short* __restrict__ qb,    // [T,DIM] bf16
    const unsigned short* __restrict__ kbuf,  // [T,DIM] bf16
    const unsigned short* __restrict__ vtb,   // [B*H][HD][SEQ] bf16
    unsigned short* __restrict__ ctxb) {      // [T,DIM] bf16
    __shared__ __align__(32) unsigned short pbuf[4][16*32];

    int lane = threadIdx.x & 31;
    int wid  = threadIdx.x >> 5;
    int wave = blockIdx.x * 4 + wid;
    int qtile = wave & 63;
    int bh    = wave >> 6;          // 0..63
    int b = bh >> 4, h = bh & 15;
    int l15 = lane & 15, hi = lane >> 4;
    int hcol = h * HD;
    int qbase = b * SEQ + qtile * 16;
    const float scale = 0.125f;     // 1/sqrt(64)

    // Q A-fragments for K-chunks c=0 (hd 0..31) and c=1 (hd 32..63)
    v16bf qf[2];
#pragma unroll
    for (int c = 0; c < 2; ++c) {
        const unsigned short* Qr =
            qb + (size_t)(qbase + l15) * DIM + hcol + c*32 + (hi ? 8 : 0);
        qf[c] = cat8(*(const v8bf*)Qr, *(const v8bf*)(Qr + 16));
    }

    v8f acc[4] = {};
    float rm[8], rl[8];
#pragma unroll
    for (int v = 0; v < 8; ++v) { rm[v] = -INFINITY; rl[v] = 0.f; }

    for (int kc = 0; kc < SEQ; kc += 32) {
        // scores: two 16x16 tiles (keys kc..+15, kc+16..+31)
        v8f s0 = {}, s1 = {};
#pragma unroll
        for (int c = 0; c < 2; ++c) {
            const unsigned short* kp0 =
                kbuf + (size_t)(b*SEQ + kc + l15) * DIM + hcol + c*32 + (hi ? 8 : 0);
            s0 = wmma_bf16(qf[c], cat8(*(const v8bf*)kp0, *(const v8bf*)(kp0 + 16)), s0);
            const unsigned short* kp1 =
                kbuf + (size_t)(b*SEQ + kc + 16 + l15) * DIM + hcol + c*32 + (hi ? 8 : 0);
            s1 = wmma_bf16(qf[c], cat8(*(const v8bf*)kp1, *(const v8bf*)(kp1 + 16)), s1);
        }

        float fac[8];
#pragma unroll
        for (int v = 0; v < 8; ++v) {
            float a0 = s0[v] * scale, a1 = s1[v] * scale;
            float m = fmaxf(a0, a1);
            for (int off = 1; off < 16; off <<= 1) m = fmaxf(m, __shfl_xor(m, off, 32));
            float mnew = fmaxf(rm[v], m);
            float p0 = __expf(a0 - mnew);
            float p1 = __expf(a1 - mnew);
            s0[v] = p0; s1[v] = p1;
            float rs = p0 + p1;
            for (int off = 1; off < 16; off <<= 1) rs += __shfl_xor(rs, off, 32);
            fac[v] = __expf(rm[v] - mnew);
            rl[v] = rl[v] * fac[v] + rs;
            rm[v] = mnew;
        }
#pragma unroll
        for (int nf = 0; nf < 4; ++nf)
#pragma unroll
            for (int v = 0; v < 8; ++v) acc[nf][v] *= fac[v];

        // C-layout P  ->  A-layout P via per-wave LDS tile
        __syncthreads();
        unsigned short* pb = pbuf[wid];
#pragma unroll
        for (int v = 0; v < 8; ++v) {
            int m = v + 8*hi;
            pb[m*32 + l15]      = f2bf(s0[v]);
            pb[m*32 + 16 + l15] = f2bf(s1[v]);
        }
        __syncthreads();
        const unsigned short* pr = pb + l15*32 + (hi ? 8 : 0);
        v16bf pf = cat8(*(const v8bf*)pr, *(const v8bf*)(pr + 16));

        // ctx += P @ V   (V head-transposed -> contiguous B-frag loads)
#pragma unroll
        for (int nf = 0; nf < 4; ++nf) {
            const unsigned short* vp =
                vtb + (((size_t)bh*HD + nf*16 + l15) << 10) + kc + (hi ? 8 : 0);
            acc[nf] = wmma_bf16(pf, cat8(*(const v8bf*)vp, *(const v8bf*)(vp + 16)), acc[nf]);
        }
    }

#pragma unroll
    for (int v = 0; v < 8; ++v) rl[v] = 1.f / rl[v];
#pragma unroll
    for (int nf = 0; nf < 4; ++nf) {
#pragma unroll
        for (int v = 0; v < 8; ++v) {
            int row = qbase + v + 8*hi;
            int col = hcol + nf*16 + l15;
            ctxb[(size_t)row * DIM + col] = f2bf(acc[nf][v] * rl[v]);
        }
    }
}

// ---------------- host launcher ----------------
extern "C" void kernel_launch(void* const* d_in, const int* in_sizes, int n_in,
                              void* d_out, int out_size, void* d_ws, size_t ws_size,
                              hipStream_t stream) {
    (void)in_sizes; (void)n_in; (void)out_size; (void)ws_size;

    const float* x  = (const float*)d_in[0];
    const float* Wr = (const float*)d_in[1];
    const float* Wq = (const float*)d_in[2];
    const float* Wk = (const float*)d_in[3];
    const float* Wv = (const float*)d_in[4];
    const float* Wo = (const float*)d_in[5];
    float* out = (float*)d_out;

    size_t off = 0;
    auto alloc = [&](size_t bytes) {
        void* p = (char*)d_ws + off;
        off += (bytes + 255) & ~(size_t)255;
        return p;
    };
    const size_t XN = (size_t)T_TOK * DIM;          // 4.19M elems
    const size_t WN = (size_t)NEXP * DIM * DIM;     // 12.58M elems

    unsigned short* xb   = (unsigned short*)alloc(XN * 2);
    unsigned short* wqb  = (unsigned short*)alloc(WN * 2);
    unsigned short* wkb  = (unsigned short*)alloc(WN * 2);
    unsigned short* wvb  = (unsigned short*)alloc(WN * 2);
    unsigned short* wob  = (unsigned short*)alloc(WN * 2);
    float* wts   = (float*)alloc((size_t)T_TOK * NEXP * 4);
    float* probs = (float*)alloc((size_t)T_TOK * NEXP * 4);
    float* psum  = (float*)alloc(NEXP * 4);
    float* cnt   = (float*)alloc(NEXP * 4);
    unsigned short* q_b  = (unsigned short*)alloc(XN * 2);
    unsigned short* k_b  = (unsigned short*)alloc(XN * 2);
    unsigned short* vtb  = (unsigned short*)alloc(XN * 2);
    unsigned short* ctxb = (unsigned short*)alloc(XN * 2);

    // 1) conversions to bf16
    cvt_bf16_kernel<<<2048, 256, 0, stream>>>(x,  xb,  (int)XN);
    cvt_bf16_kernel<<<4096, 256, 0, stream>>>(Wq, wqb, (int)WN);
    cvt_bf16_kernel<<<4096, 256, 0, stream>>>(Wk, wkb, (int)WN);
    cvt_bf16_kernel<<<4096, 256, 0, stream>>>(Wv, wvb, (int)WN);
    cvt_bf16_kernel<<<4096, 256, 0, stream>>>(Wo, wob, (int)WN);

    // 2) router + deterministic LB scalar
    router_kernel<<<T_TOK/4, 128, 0, stream>>>(x, Wr, wts, probs);
    lb_reduce_kernel<<<NEXP, 256, 0, stream>>>(probs, wts, psum, cnt);
    lb_final_kernel<<<1, 1, 0, stream>>>(psum, cnt, out + XN);

    // 3) MoE QKV projections (WMMA); V stored head-transposed
    moe_gemm_kernel<<<256, 256, 0, stream>>>(xb, wqb, wts, q_b, 1);
    moe_gemm_kernel<<<256, 256, 0, stream>>>(xb, wkb, wts, k_b, 1);
    moe_gemm_kernel<<<256, 256, 0, stream>>>(xb, wvb, wts, vtb, 2);

    // 4) flash attention (WMMA + fp32 online softmax) -> bf16 ctx
    attn_kernel<<<1024, 128, 0, stream>>>(q_b, k_b, vtb, ctxb);

    // 5) MoE output projection -> fp32 final output
    moe_gemm_kernel<<<256, 256, 0, stream>>>(ctxb, wob, wts, out, 0);
}
